// MoE_68436008894523
// MI455X (gfx1250) — matrix operand order repaired and checked by reading, not compiled, not run on hardware
//
#include <hip/hip_runtime.h>
#include <hip/hip_bf16.h>
#include <math.h>

// Problem constants (match reference): B=2, S=2048 -> T=4096, H=1024, E=8, F=4096
#define T_TOKENS 4096
#define H_DIM    1024
#define E_EXP    8
#define F_DIM    4096
#define LN_EPS   1e-5f
#define MTILE    32          // tokens per block (two 16-row WMMA subtiles)

typedef __attribute__((ext_vector_type(16))) __bf16 v16bf;
typedef __attribute__((ext_vector_type(8)))  float  v8f;

__device__ __forceinline__ float gelu_tanh(float v) {
  const float c = 0.7978845608028654f;  // sqrt(2/pi)
  return 0.5f * v * (1.0f + tanhf(c * (v + 0.044715f * v * v * v)));
}

// ---------------------------------------------------------------------------
// Kernel 1: gating. One wave (32 lanes) per token.
//   logits = x[t] . Wg  (H x E), top-2, softmax over the 2, scale by alpha.
//   Compact (token, gate) into per-expert lists with atomics.
// ---------------------------------------------------------------------------
__global__ __launch_bounds__(256) void moe_gate_kernel(
    const float* __restrict__ x, const float* __restrict__ Wg,
    const float* __restrict__ alpha, int* __restrict__ cnt,
    int* __restrict__ tok_list, float* __restrict__ gate_list) {
  int t    = blockIdx.x * 8 + (threadIdx.x >> 5);
  int lane = threadIdx.x & 31;
  const float* xr = x + (size_t)t * H_DIM;

  float acc[E_EXP];
#pragma unroll
  for (int e = 0; e < E_EXP; ++e) acc[e] = 0.f;

  for (int i = 0; i < H_DIM / 32; ++i) {
    int h = lane + i * 32;
    float xv = xr[h];
    const float* wr = Wg + (size_t)h * E_EXP;
#pragma unroll
    for (int e = 0; e < E_EXP; ++e) acc[e] += xv * wr[e];
  }
#pragma unroll
  for (int e = 0; e < E_EXP; ++e) {
#pragma unroll
    for (int off = 16; off > 0; off >>= 1) acc[e] += __shfl_xor(acc[e], off, 32);
  }

  if (lane == 0) {
    // top-1
    int b0 = 0; float v0 = acc[0];
#pragma unroll
    for (int e = 1; e < E_EXP; ++e) { if (acc[e] > v0) { v0 = acc[e]; b0 = e; } }
    // top-2
    int b1 = (b0 == 0) ? 1 : 0; float v1 = acc[b1];
#pragma unroll
    for (int e = 0; e < E_EXP; ++e) {
      if (e != b0 && acc[e] > v1) { v1 = acc[e]; b1 = e; }
    }
    float p0 = 1.f / (1.f + expf(v1 - v0));   // softmax over {v0, v1}
    float p1 = 1.f - p0;

    int pos0 = atomicAdd(&cnt[b0], 1);
    tok_list [b0 * T_TOKENS + pos0] = t;
    gate_list[b0 * T_TOKENS + pos0] = p0 * alpha[b0];
    int pos1 = atomicAdd(&cnt[b1], 1);
    tok_list [b1 * T_TOKENS + pos1] = t;
    gate_list[b1 * T_TOKENS + pos1] = p1 * alpha[b1];
  }
}

// ---------------------------------------------------------------------------
// Kernel 2: fused per-expert FFN on a 32-token tile.
//   256 threads = 8 waves. LN(x) -> bf16 in LDS, then for each 128-wide F
//   chunk: fc1 (WMMA bf16, 2 M-subtiles sharing each B fragment) + gelu ->
//   LDS, fc2 partial-K (WMMA bf16) into persistent accumulators (each wave
//   owns 128 of the 1024 H columns x 2 M-subtiles).
//   Epilogue: +bias2, *gate, atomic-add to out.
// ---------------------------------------------------------------------------
__global__ __launch_bounds__(256) void moe_ffn_kernel(
    const float* __restrict__ x,
    const float* __restrict__ ln_w,  const float* __restrict__ ln_b,
    const float* __restrict__ fc1_w, const float* __restrict__ fc1_b,
    const float* __restrict__ fc2_w, const float* __restrict__ fc2_b,
    const int* __restrict__ cnt, const int* __restrict__ tok_list,
    const float* __restrict__ gate_list, float* __restrict__ out) {
  int e     = blockIdx.y;
  int n_tok = cnt[e];
  int tile0 = blockIdx.x * MTILE;
  if (tile0 >= n_tok) return;            // block-uniform: EXEC stays all-1 below

  __shared__ __bf16 x_lds[MTILE][H_DIM + 8];   // LayerNormed activations (bf16)
  __shared__ __bf16 a_lds[MTILE][128 + 8];     // gelu(fc1) chunk (bf16)
  __shared__ int    toks[MTILE];
  __shared__ float  gates[MTILE];

  int tid = threadIdx.x;
  if (tid < MTILE) {
    int idx    = tile0 + tid;
    bool valid = idx < n_tok;
    toks[tid]  = tok_list [e * T_TOKENS + (valid ? idx : tile0)];
    gates[tid] = valid ? gate_list[e * T_TOKENS + idx] : 0.f;
  }
  __syncthreads();

  // ---- LayerNorm: token m handled by 8 threads (g = lane-in-group) ----
  {
    int m = tid >> 3, g = tid & 7;
    const float* xr = x + (size_t)toks[m] * H_DIM;
    float s = 0.f, s2 = 0.f;
    for (int j = 0; j < H_DIM / 8; ++j) {
      float v = xr[g + 8 * j];
      s += v; s2 += v * v;
    }
#pragma unroll
    for (int off = 4; off > 0; off >>= 1) {
      s  += __shfl_xor(s,  off, 8);
      s2 += __shfl_xor(s2, off, 8);
    }
    float mu   = s  * (1.f / H_DIM);
    float var  = s2 * (1.f / H_DIM) - mu * mu;
    float rstd = rsqrtf(var + LN_EPS);
    const float* lw = ln_w + (size_t)e * H_DIM;
    const float* lb = ln_b + (size_t)e * H_DIM;
    for (int j = 0; j < H_DIM / 8; ++j) {
      int h = g + 8 * j;
      x_lds[m][h] = (__bf16)((xr[h] - mu) * rstd * lw[h] + lb[h]);
    }
  }
  __syncthreads();

  int w = tid >> 5, lane = tid & 31, half = lane >> 4, nl = lane & 15;

  const float* W1e = fc1_w + (size_t)e * H_DIM * F_DIM;
  const float* W2e = fc2_w + (size_t)e * F_DIM * H_DIM;
  const float* b1e = fc1_b + (size_t)e * F_DIM;

  v8f acc2[2][8];
#pragma unroll
  for (int ms = 0; ms < 2; ++ms)
#pragma unroll
    for (int i = 0; i < 8; ++i) acc2[ms][i] = (v8f){};

  for (int f0 = 0; f0 < F_DIM; f0 += 128) {
    // ---------- fc1: this wave computes columns [f0 + 16w, +16) ----------
    int n0 = f0 + w * 16;
    v8f acc1[2] = {(v8f){}, (v8f){}};
    for (int k0 = 0; k0 < H_DIM; k0 += 32) {
      v16bf a0, a1, b;
      // A fragment (16x32 bf16): lanes 0-15 -> K {0..7,16..23}; 16-31 -> K {8..15,24..31}
#pragma unroll
      for (int j = 0; j < 16; ++j) {
        int kk = (j & 7) + ((j >> 3) << 4) + half * 8;
        a0[j] = x_lds[nl][k0 + kk];
        a1[j] = x_lds[nl + 16][k0 + kk];
      }
      // B fragment (32x16 bf16): lane = N column; lanes 0-15 -> K {0..7,16..23}; 16-31 -> K {8..15,24..31}
#pragma unroll
      for (int j = 0; j < 16; ++j) {
        int kk = (j & 7) + ((j >> 3) << 4) + half * 8;
        b[j] = (__bf16)W1e[(size_t)(k0 + kk) * F_DIM + n0 + nl];
      }
      if (k0 + 32 < H_DIM)
        __builtin_prefetch(&W1e[(size_t)(k0 + 32 + half * 16) * F_DIM + n0 + nl], 0, 1);
      acc1[0] = __builtin_amdgcn_wmma_f32_16x16x32_bf16(
          false, a0, false, b, (short)0, acc1[0], false, false);
      acc1[1] = __builtin_amdgcn_wmma_f32_16x16x32_bf16(
          false, a1, false, b, (short)0, acc1[1], false, false);
    }
    // bias + tanh-gelu -> bf16 activation tile
#pragma unroll
    for (int ms = 0; ms < 2; ++ms) {
#pragma unroll
      for (int r = 0; r < 8; ++r) {
        int m = ms * 16 + r + 8 * half;          // C/D layout: M = r + 8*half
        float v = acc1[ms][r] + b1e[n0 + nl];
        a_lds[m][w * 16 + nl] = (__bf16)gelu_tanh(v);
      }
    }
    __syncthreads();

    // ---------- fc2 partial-K: this wave owns H columns [128w, +128) -----
#pragma unroll
    for (int kk0 = 0; kk0 < 4; ++kk0) {
      v16bf a0, a1;
#pragma unroll
      for (int j = 0; j < 16; ++j) {
        int kk = (j & 7) + ((j >> 3) << 4) + half * 8;
        a0[j] = a_lds[nl][kk0 * 32 + kk];
        a1[j] = a_lds[nl + 16][kk0 * 32 + kk];
      }
#pragma unroll
      for (int nt = 0; nt < 8; ++nt) {
        v16bf b;
#pragma unroll
        for (int j = 0; j < 16; ++j) {
          int kk = (j & 7) + ((j >> 3) << 4) + half * 8;
          b[j] = (__bf16)W2e[(size_t)(f0 + kk0 * 32 + kk) * H_DIM + w * 128 + nt * 16 + nl];
        }
        acc2[0][nt] = __builtin_amdgcn_wmma_f32_16x16x32_bf16(
            false, a0, false, b, (short)0, acc2[0][nt], false, false);
        acc2[1][nt] = __builtin_amdgcn_wmma_f32_16x16x32_bf16(
            false, a1, false, b, (short)0, acc2[1][nt], false, false);
      }
    }
    __syncthreads();   // a_lds reused next chunk
  }

  // ---------- epilogue: +bias2, *gate, scatter-accumulate ----------
  const float* b2e = fc2_b + (size_t)e * H_DIM;
#pragma unroll
  for (int ms = 0; ms < 2; ++ms) {
#pragma unroll
    for (int nt = 0; nt < 8; ++nt) {
#pragma unroll
      for (int r = 0; r < 8; ++r) {
        int m = ms * 16 + r + 8 * half;
        int h = w * 128 + nt * 16 + nl;
        float v = (acc2[ms][nt][r] + b2e[h]) * gates[m];
        atomicAdd(&out[(size_t)toks[m] * H_DIM + h], v);
      }
    }
  }
}

// ---------------------------------------------------------------------------
extern "C" void kernel_launch(void* const* d_in, const int* in_sizes, int n_in,
                              void* d_out, int out_size, void* d_ws, size_t ws_size,
                              hipStream_t stream) {
  (void)in_sizes; (void)n_in; (void)ws_size;
  const float* x     = (const float*)d_in[0];
  const float* Wg    = (const float*)d_in[1];
  const float* alpha = (const float*)d_in[2];
  const float* ln_w  = (const float*)d_in[3];
  const float* ln_b  = (const float*)d_in[4];
  const float* fc1_w = (const float*)d_in[5];
  const float* fc1_b = (const float*)d_in[6];
  const float* fc2_w = (const float*)d_in[7];
  const float* fc2_b = (const float*)d_in[8];
  float* out = (float*)d_out;

  // Workspace layout: [cnt: 256B] [tok_list: E*T ints] [gate_list: E*T floats]
  char*  ws        = (char*)d_ws;
  int*   cnt       = (int*)ws;
  int*   tok_list  = (int*)(ws + 256);
  float* gate_list = (float*)(ws + 256 + sizeof(int) * (size_t)E_EXP * T_TOKENS);

  hipMemsetAsync(d_out, 0, (size_t)out_size * sizeof(float), stream);
  hipMemsetAsync(cnt, 0, 256, stream);

  moe_gate_kernel<<<T_TOKENS / 8, 256, 0, stream>>>(x, Wg, alpha, cnt,
                                                    tok_list, gate_list);
  moe_ffn_kernel<<<dim3(T_TOKENS / MTILE, E_EXP), 256, 0, stream>>>(
      x, ln_w, ln_b, fc1_w, fc1_b, fc2_w, fc2_b, cnt, tok_list, gate_list, out);
}
